// PWCDecoder_from_img_18863496364615
// MI455X (gfx1250) — matrix operand order repaired and checked by reading, not compiled, not run on hardware
//
#include <hip/hip_runtime.h>

typedef __attribute__((ext_vector_type(16))) _Float16 v16h;
typedef __attribute__((ext_vector_type(8)))  _Float16 v8h;
typedef __attribute__((ext_vector_type(8)))  float    v8f;
typedef __attribute__((ext_vector_type(4)))  float    v4f;

// ---------------------------------------------------------------------------
// Weight packing: [Cout,Cin,k,k] f32 -> f16 [Mpad][k*k][Cpad], zero padded.
// K dimension enumerates (p, ci) with p = ky*k+kx (position-major), so the
// conv kernel's inner loop over channels needs no div/mod.
// ---------------------------------------------------------------------------
__global__ void pack_weights(const float* __restrict__ w, _Float16* __restrict__ out,
                             int Cout, int Cin, int ksz, int Cpad, int total) {
    int idx = blockIdx.x * blockDim.x + threadIdx.x;
    if (idx >= total) return;
    int Kpad = ksz * ksz * Cpad;
    int m  = idx / Kpad;
    int r  = idx - m * Kpad;
    int p  = r / Cpad;
    int ci = r - p * Cpad;
    int ky = p / ksz;
    int kx = p - ky * ksz;
    float v = (m < Cout && ci < Cin)
            ? w[((size_t)m * Cin + ci) * (ksz * ksz) + ky * ksz + kx] : 0.0f;
    out[idx] = (_Float16)v;
}

// ---------------------------------------------------------------------------
// NHWC implicit-GEMM conv via v_wmma_f32_16x16x32_f16.
// One wave = 32 output channels x 16 output pixels (two M-tiles sharing one
// B fragment -> 2 WMMAs per K-window), 4 waves/block.
// A: two 128-bit f16 weight loads per M-tile per K-window.
// B: four 128-bit f32 loads (16 consecutive channels) per K-window, reused.
// Loops: ky,kx outermost (boundary mask once per position), channels inner
// in steps of 32. Second M-tile guarded by wave-uniform hasM1 (EXEC stays
// all-ones for every executed WMMA).
// Epilogue: +bias, optional LeakyReLU(0.1), write channel slice
// [outC0, outC0+Cout) of an outCtot-channel NHWC tensor.
// ---------------------------------------------------------------------------
__global__ __launch_bounds__(128) void conv_wmma(
    const float* __restrict__ in, int inCtot, int H, int W,
    const _Float16* __restrict__ wgt, int Cpad, int Kpad, int Mpad,
    const float* __restrict__ bias,
    float* __restrict__ out, int outCtot, int outC0, int Cout, int Ho, int Wo,
    int B, int ksz, int stride, int pad, int dil, int act)
{
    const int  lane = threadIdx.x & 31;
    const int  wave = threadIdx.x >> 5;
    const int  col  = lane & 15;
    const bool hi   = lane >= 16;
    const int  hwo  = Ho * Wo;
    const int  Ntot = B * hwo;
    const int  n0   = (blockIdx.x * 4 + wave) * 16;
    const int  m0   = blockIdx.y * 32;
    if (n0 >= Ntot) return;                    // wave-uniform
    const bool hasM1 = (m0 + 16) < Mpad;       // wave-uniform

    int n = n0 + col;
    const bool nv = n < Ntot;
    int nc = nv ? n : (Ntot - 1);
    int b  = nc / hwo;
    int r0 = nc - b * hwo;
    int oy = r0 / Wo;
    int ox = r0 - oy * Wo;
    const int iy0 = oy * stride - pad;
    const int ix0 = ox * stride - pad;

    const _Float16* wrow0 = wgt + (size_t)(m0 + col) * Kpad + (hi ? 16 : 0);
    const _Float16* wrow1 = wrow0 + (size_t)16 * Kpad;
    const int       koff  = hi ? 16 : 0;

    v8f acc0 = {};
    v8f acc1 = {};
    int kwin = 0;
    for (int ky = 0; ky < ksz; ++ky) {
        const int  iy = iy0 + ky * dil;
        const bool yv = (unsigned)iy < (unsigned)H;
        for (int kx = 0; kx < ksz; ++kx) {
            const int  ix = ix0 + kx * dil;
            const bool pvalid = nv && yv && ((unsigned)ix < (unsigned)W);
            const float* src = in + ((size_t)(b * H + iy) * W + ix) * inCtot + koff;
            for (int c0 = 0; c0 < Cpad; c0 += 32, kwin += 32) {
                v16h bm;
                if (pvalid) {
                    const v4f* sp = (const v4f*)(src + c0);
                    v4f f0 = sp[0], f1 = sp[1], f2 = sp[2], f3 = sp[3];
#pragma unroll
                    for (int h = 0; h < 4; ++h) {
                        bm[h]      = (_Float16)f0[h];
                        bm[h + 4]  = (_Float16)f1[h];
                        bm[h + 8]  = (_Float16)f2[h];
                        bm[h + 12] = (_Float16)f3[h];
                    }
                } else {
#pragma unroll
                    for (int h = 0; h < 16; ++h) bm[h] = (_Float16)0.0f;
                }

                {
                    v8h alo = *(const v8h*)(wrow0 + kwin);
                    v8h ahi = *(const v8h*)(wrow0 + kwin + 8);
                    v16h a;
#pragma unroll
                    for (int h = 0; h < 8; ++h) { a[h] = alo[h]; a[h + 8] = ahi[h]; }
                    acc0 = __builtin_amdgcn_wmma_f32_16x16x32_f16(false, a, false, bm,
                                                                  (short)0, acc0, false, false);
                }
                if (hasM1) {
                    v8h alo = *(const v8h*)(wrow1 + kwin);
                    v8h ahi = *(const v8h*)(wrow1 + kwin + 8);
                    v16h a;
#pragma unroll
                    for (int h = 0; h < 8; ++h) { a[h] = alo[h]; a[h + 8] = ahi[h]; }
                    acc1 = __builtin_amdgcn_wmma_f32_16x16x32_f16(false, a, false, bm,
                                                                  (short)0, acc1, false, false);
                }
            }
        }
    }

    float* opix = out + ((size_t)(b * Ho + oy) * Wo + ox) * outCtot + outC0;
    const int mb0 = m0 + (hi ? 8 : 0);
#pragma unroll
    for (int r = 0; r < 8; ++r) {
        int m = mb0 + r;
        if (m < Cout && nv) {
            float v = acc0[r] + bias[m];
            if (act) v = (v >= 0.0f) ? v : 0.1f * v;
            opix[m] = v;
        }
    }
    if (hasM1) {
        const int mb1 = mb0 + 16;
#pragma unroll
        for (int r = 0; r < 8; ++r) {
            int m = mb1 + r;
            if (m < Cout && nv) {
                float v = acc1[r] + bias[m];
                if (act) v = (v >= 0.0f) ? v : 0.1f * v;
                opix[m] = v;
            }
        }
    }
}

// ---------------------------------------------------------------------------
// Correlation (81 displacements, mean over C, fused LeakyReLU), NHWC.
// Writes channels [0,81) of an outCtot-channel tensor. C multiple of 4.
// ---------------------------------------------------------------------------
__global__ void corr_kernel(const float* __restrict__ x1, const float* __restrict__ x2,
                            float* __restrict__ out, int outCtot,
                            int B, int C, int H, int W) {
    int idx = blockIdx.x * blockDim.x + threadIdx.x;
    int total = B * H * W * 81;
    if (idx >= total) return;
    int d = idx % 81; int t = idx / 81;
    int x = t % W;    t /= W;
    int y = t % H;    int b = t / H;
    int dq = d / 9;
    int dy = dq - 4, dx = (d - dq * 9) - 4;
    int y2 = y + dy, x2c = x + dx;
    float s = 0.0f;
    if ((unsigned)y2 < (unsigned)H && (unsigned)x2c < (unsigned)W) {
        const v4f* p1 = (const v4f*)(x1 + ((size_t)(b * H + y)  * W + x)   * C);
        const v4f* p2 = (const v4f*)(x2 + ((size_t)(b * H + y2) * W + x2c) * C);
        for (int c = 0; c < C / 4; ++c) {
            v4f a = p1[c], bb = p2[c];
            s += a[0] * bb[0] + a[1] * bb[1] + a[2] * bb[2] + a[3] * bb[3];
        }
    }
    s *= (1.0f / (float)C);
    s = (s >= 0.0f) ? s : 0.1f * s;
    out[((size_t)(b * H + y) * W + x) * outCtot + d] = s;
}

// ---------------------------------------------------------------------------
// Backward warp (bilinear, zeros pad, validity mask), NHWC, C multiple of 4.
// flow is NHWC [B,H,W,2].
// ---------------------------------------------------------------------------
__global__ void warp_kernel(const float* __restrict__ x, const float* __restrict__ flow,
                            float* __restrict__ out, int B, int C, int H, int W) {
    int idx = blockIdx.x * blockDim.x + threadIdx.x;
    int total = B * H * W;
    if (idx >= total) return;
    int xc = idx % W; int t = idx / W;
    int y  = t % H;   int b = t / H;
    size_t pix = (size_t)(b * H + y) * W + xc;
    float gx = (float)xc + flow[pix * 2 + 0];
    float gy = (float)y  + flow[pix * 2 + 1];
    float x0f = floorf(gx), y0f = floorf(gy);
    float wx1 = gx - x0f, wx0 = 1.0f - wx1;
    float wy1 = gy - y0f, wy0 = 1.0f - wy1;
    int xi[2], yi[2]; float mx[2], my[2];
#pragma unroll
    for (int i = 0; i < 2; ++i) {
        float xf = x0f + (float)i;
        float yf = y0f + (float)i;
        xi[i] = (int)fminf(fmaxf(xf, 0.0f), (float)(W - 1));
        yi[i] = (int)fminf(fmaxf(yf, 0.0f), (float)(H - 1));
        mx[i] = (xf >= 0.0f && xf <= (float)(W - 1)) ? 1.0f : 0.0f;
        my[i] = (yf >= 0.0f && yf <= (float)(H - 1)) ? 1.0f : 0.0f;
    }
    float w00 = wx0 * wy0 * mx[0] * my[0];
    float w01 = wx1 * wy0 * mx[1] * my[0];
    float w10 = wx0 * wy1 * mx[0] * my[1];
    float w11 = wx1 * wy1 * mx[1] * my[1];
    float mval = w00 + w01 + w10 + w11;
    float mask = (mval < 0.9999f) ? 0.0f : 1.0f;
    const float* xb = x + (size_t)b * H * W * C;
    const v4f* p00 = (const v4f*)(xb + ((size_t)yi[0] * W + xi[0]) * C);
    const v4f* p01 = (const v4f*)(xb + ((size_t)yi[0] * W + xi[1]) * C);
    const v4f* p10 = (const v4f*)(xb + ((size_t)yi[1] * W + xi[0]) * C);
    const v4f* p11 = (const v4f*)(xb + ((size_t)yi[1] * W + xi[1]) * C);
    v4f* op = (v4f*)(out + pix * C);
    for (int c = 0; c < C / 4; ++c) {
        v4f v = (p00[c] * w00 + p01[c] * w01 + p10[c] * w10 + p11[c] * w11) * mask;
        op[c] = v;
    }
}

// ---------------------------------------------------------------------------
// 2x bilinear upsample (align_corners=True) with scale, NHWC [B,H,W,2]
// ---------------------------------------------------------------------------
__global__ void upflow_kernel(const float* __restrict__ in, float* __restrict__ out,
                              int B, int H, int W, float scale) {
    int H2 = 2 * H, W2 = 2 * W;
    int idx = blockIdx.x * blockDim.x + threadIdx.x;
    int total = B * H2 * W2 * 2;
    if (idx >= total) return;
    int c = idx % 2;  int t = idx / 2;
    int x = t % W2;   t /= W2;
    int y = t % H2;   int b = t / H2;
    float ys = (float)y * (float)(H - 1) / (float)(H2 - 1);
    float xs = (float)x * (float)(W - 1) / (float)(W2 - 1);
    int y0 = (int)floorf(ys); if (y0 > H - 2) y0 = H - 2; if (y0 < 0) y0 = 0;
    int x0 = (int)floorf(xs); if (x0 > W - 2) x0 = W - 2; if (x0 < 0) x0 = 0;
    float wy = ys - (float)y0, wx = xs - (float)x0;
    const float* p = in + (size_t)b * H * W * 2;
    float v00 = p[((size_t)y0 * W + x0) * 2 + c];
    float v01 = p[((size_t)y0 * W + x0 + 1) * 2 + c];
    float v10 = p[((size_t)(y0 + 1) * W + x0) * 2 + c];
    float v11 = p[((size_t)(y0 + 1) * W + x0 + 1) * 2 + c];
    float t0 = v00 * (1.0f - wy) + v10 * wy;
    float t1 = v01 * (1.0f - wy) + v11 * wy;
    out[idx] = (t0 * (1.0f - wx) + t1 * wx) * scale;
}

// ---------------------------------------------------------------------------
// Layout transforms + small utilities
// ---------------------------------------------------------------------------
__global__ void nchw_to_nhwc(const float* __restrict__ src, float* __restrict__ dst,
                             int B, int C, int Ctot, int H, int W) {
    int idx = blockIdx.x * blockDim.x + threadIdx.x;
    int total = B * H * W * Ctot;
    if (idx >= total) return;
    int c = idx % Ctot; int t = idx / Ctot;
    int x = t % W;      t /= W;
    int y = t % H;      int b = t / H;
    dst[idx] = (c < C) ? src[(((size_t)b * C + c) * H + y) * W + x] : 0.0f;
}

__global__ void nhwc_to_nchw(const float* __restrict__ src, float* __restrict__ dst,
                             int B, int C, int H, int W) {
    int idx = blockIdx.x * blockDim.x + threadIdx.x;
    int total = B * C * H * W;
    if (idx >= total) return;
    int x = idx % W; int t = idx / W;
    int y = t % H;   t /= H;
    int c = t % C;   int b = t / C;
    dst[idx] = src[((size_t)(b * H + y) * W + x) * C + c];
}

__global__ void copy_ch(const float* __restrict__ src, int sCtot, int sC0,
                        float* __restrict__ dst, int dCtot, int dC0,
                        int B, int C, int H, int W) {
    int idx = blockIdx.x * blockDim.x + threadIdx.x;
    int total = B * H * W * C;
    if (idx >= total) return;
    int c = idx % C; int t = idx / C;
    int x = t % W;   t /= W;
    int y = t % H;   int b = t / H;
    size_t pix = (size_t)(b * H + y) * W + x;
    dst[pix * dCtot + dC0 + c] = src[pix * sCtot + sC0 + c];
}

__global__ void add_inplace(float* __restrict__ a, const float* __restrict__ b, int n) {
    int i = blockIdx.x * blockDim.x + threadIdx.x;
    if (i < n) a[i] += b[i];
}

__global__ void fill_zero(float* __restrict__ p, int n) {
    int i = blockIdx.x * blockDim.x + threadIdx.x;
    if (i < n) p[i] = 0.0f;
}

// ---------------------------------------------------------------------------
// Host orchestration
// ---------------------------------------------------------------------------
extern "C" void kernel_launch(void* const* d_in, const int* in_sizes, int n_in,
                              void* d_out, int out_size, void* d_ws, size_t ws_size,
                              hipStream_t stream) {
    (void)in_sizes; (void)n_in; (void)out_size; (void)ws_size;
    const int B = 4;

    // ---- bump allocator over d_ws ----
    char* base = (char*)d_ws;
    size_t off = 0;
    auto alloc = [&](size_t nbytes) -> void* {
        off = (off + 255) & ~(size_t)255;
        void* p = base + off;
        off += nbytes;
        return p;
    };
    auto allocF = [&](size_t nfloats) -> float* { return (float*)alloc(nfloats * sizeof(float)); };

    // ---- layer specs (d_in order: im1, im2, ctx, est, feat, px — pytree sorted keys) ----
    const int FEAT_CIN[18]  = {3,16,16,16,32,32,32,64,64,64,96,96,96,128,128,128,196,196};
    const int FEAT_COUT[18] = {16,16,16,32,32,32,64,64,64,96,96,96,128,128,128,196,196,196};
    const int FEAT_S[18]    = {2,1,1,2,1,1,2,1,1,2,1,1,2,1,1,2,1,1};
    const int EST_CIN[6]  = {115,128,256,224,160,96};
    const int EST_COUT[6] = {128,128,96,64,32,2};
    const int CTX_CIN[7]  = {34,128,128,128,96,64,32};
    const int CTX_COUT[7] = {128,128,128,96,64,32,2};
    const int CTX_D[7]    = {1,2,4,8,16,1,1};
    const int PXC[5]      = {196,128,96,64,32};

    struct PW { const _Float16* w; const float* bias; int cout, k, Cpad, Kpad, Mpad; };
    auto prep = [&](int widx, int cin, int cout, int k) -> PW {
        PW pw; pw.cout = cout; pw.k = k;
        pw.Cpad = (cin + 31) & ~31;
        pw.Kpad = k * k * pw.Cpad;
        pw.Mpad = (cout + 15) & ~15;
        int total = pw.Mpad * pw.Kpad;
        _Float16* wp = (_Float16*)alloc(sizeof(_Float16) * (size_t)total);
        pack_weights<<<(total + 255) / 256, 256, 0, stream>>>(
            (const float*)d_in[widx], wp, cout, cin, k, pw.Cpad, total);
        pw.w = wp; pw.bias = (const float*)d_in[widx + 1];
        return pw;
    };

    PW ctxW[7], estW[6], featW[18], pxW[5];
    for (int i = 0; i < 7;  ++i) ctxW[i]  = prep(2  + 2 * i, CTX_CIN[i],  CTX_COUT[i], 3);
    for (int i = 0; i < 6;  ++i) estW[i]  = prep(16 + 2 * i, EST_CIN[i],  EST_COUT[i], 3);
    for (int i = 0; i < 18; ++i) featW[i] = prep(28 + 2 * i, FEAT_CIN[i], FEAT_COUT[i], 3);
    for (int i = 0; i < 5;  ++i) pxW[i]   = prep(64 + 2 * i, PXC[i], 32, 1);

    auto conv = [&](const float* in, int inCtot, int H, int W,
                    const PW& pw, float* outp, int outCtot, int outC0,
                    int Ho, int Wo, int stride, int dil, int act) {
        int Ntot = B * Ho * Wo;
        int ntiles = (Ntot + 15) / 16;
        dim3 grid((ntiles + 3) / 4, (pw.Mpad + 31) / 32);
        int pad = (pw.k - 1) * dil / 2;
        conv_wmma<<<grid, 128, 0, stream>>>(in, inCtot, H, W,
            pw.w, pw.Cpad, pw.Kpad, pw.Mpad, pw.bias,
            outp, outCtot, outC0, pw.cout, Ho, Wo,
            B, pw.k, stride, pad, dil, act);
    };
    auto eg = [&](int n) { return dim3((unsigned)((n + 255) / 256)); };

    // ---- buffers (all NHWC; Ctot multiples of 4) ----
    const int LC[5] = {196,128,96,64,32};
    const int LH[5] = {4,8,16,32,64};
    const int LW[5] = {13,26,52,104,208};

    float* im1n = allocF((size_t)B * 256 * 832 * 4);
    float* im2n = allocF((size_t)B * 256 * 832 * 4);
    float* pa = allocF((size_t)B * 128 * 416 * 16);
    float* pb = allocF((size_t)B * 128 * 416 * 16);
    float* p1f[5]; float* p2f[5];
    for (int l = 0; l < 5; ++l) {
        size_t sz = (size_t)B * LH[l] * LW[l] * LC[l];
        p1f[l] = allocF(sz); p2f[l] = allocF(sz);
    }

    const size_t PIXmax = (size_t)B * 64 * 208;      // 53248
    float* est_in  = allocF(PIXmax * 116);           // 81 corr | 32 px | 2 flow | 1 pad
    float* catA    = allocF(PIXmax * 256);
    float* catB    = allocF(PIXmax * 224);
    float* catC    = allocF(PIXmax * 160);
    float* catD    = allocF(PIXmax * 96);
    float* flowRes = allocF(PIXmax * 2);
    float* ctx_in  = allocF(PIXmax * 36);
    float* ctxP0   = allocF(PIXmax * 128);
    float* ctxP1   = allocF(PIXmax * 128);
    float* ctx_out = allocF(PIXmax * 2);
    float* flowA   = allocF(PIXmax * 2);
    float* flowB   = allocF(PIXmax * 2);
    float* x2w     = allocF(PIXmax * 64);
    (void)allocF(1024);                              // spill guard for padded-channel reads

    // ---- NCHW -> NHWC for input images (pad 3->4 channels with zeros) ----
    {
        int n = B * 256 * 832 * 4;
        nchw_to_nhwc<<<eg(n), 256, 0, stream>>>((const float*)d_in[0], im1n, B, 3, 4, 256, 832);
        nchw_to_nhwc<<<eg(n), 256, 0, stream>>>((const float*)d_in[1], im2n, B, 3, 4, 256, 832);
    }

    // ---- feature pyramids for both images ----
    auto runPyr = [&](const float* im, float** lv) {
        const float* cur = im; int H = 256, W = 832; int Ctot = 4;
        for (int i = 0; i < 18; ++i) {
            int s = FEAT_S[i];
            int Ho = (s == 2) ? H / 2 : H;
            int Wo = (s == 2) ? W / 2 : W;
            float* dst;
            if      (i == 5)  dst = lv[4];
            else if (i == 8)  dst = lv[3];
            else if (i == 11) dst = lv[2];
            else if (i == 14) dst = lv[1];
            else if (i == 17) dst = lv[0];
            else dst = (cur == pa) ? pb : pa;
            conv(cur, Ctot, H, W, featW[i], dst, FEAT_COUT[i], 0, Ho, Wo, s, 1, 1);
            cur = dst; H = Ho; W = Wo; Ctot = FEAT_COUT[i];
        }
    };
    runPyr(im1n, p1f);
    runPyr(im2n, p2f);

    // ---- coarse-to-fine decode ----
    float* flow = flowA; float* flowN = flowB;
    for (int l = 0; l < 5; ++l) {
        int C = LC[l], H = LH[l], W = LW[l];
        int npix = B * H * W;
        int HW2  = npix * 2;
        const float* x2wp;
        if (l == 0) {
            fill_zero<<<eg(HW2), 256, 0, stream>>>(flow, HW2);
            x2wp = p2f[0];
        } else {
            upflow_kernel<<<eg(HW2), 256, 0, stream>>>(flow, flowN, B, LH[l-1], LW[l-1], 2.0f);
            float* t = flow; flow = flowN; flowN = t;
            warp_kernel<<<eg(npix), 256, 0, stream>>>(p2f[l], flow, x2w, B, C, H, W);
            x2wp = x2w;
        }
        // correlation (+LReLU) into est_in channels [0,81)
        corr_kernel<<<eg(npix * 81), 256, 0, stream>>>(p1f[l], x2wp, est_in, 116, B, C, H, W);
        // 1x1 px conv into est_in channels [81,113)
        conv(p1f[l], C, H, W, pxW[l], est_in, 116, 81, H, W, 1, 1, 1);
        // flow into est_in channels [113,115)
        copy_ch<<<eg(HW2), 256, 0, stream>>>(flow, 2, 0, est_in, 116, 113, B, 2, H, W);

        // flow estimator (dense-connected)
        conv(est_in, 116, H, W, estW[0], catA, 256, 0,   H, W, 1, 1, 1);  // x1
        conv(catA,   256, H, W, estW[1], catA, 256, 128, H, W, 1, 1, 1);  // x2 (reads ch[0,128))
        copy_ch<<<eg(npix * 128), 256, 0, stream>>>(catA, 256, 128, catB, 224, 0, B, 128, H, W);
        conv(catA,   256, H, W, estW[2], catB, 224, 128, H, W, 1, 1, 1);  // x3
        copy_ch<<<eg(npix * 96), 256, 0, stream>>>(catB, 224, 128, catC, 160, 0, B, 96, H, W);
        conv(catB,   224, H, W, estW[3], catC, 160, 96,  H, W, 1, 1, 1);  // x4
        copy_ch<<<eg(npix * 64), 256, 0, stream>>>(catC, 160, 96, catD, 96, 0, B, 64, H, W);
        conv(catC,   160, H, W, estW[4], catD, 96,  64,  H, W, 1, 1, 1);  // x5 (x_intm)
        conv(catD,   96,  H, W, estW[5], flowRes, 2, 0,  H, W, 1, 1, 0);  // flow residual
        add_inplace<<<eg(HW2), 256, 0, stream>>>(flow, flowRes, HW2);

        // context net (dilated) on [x_intm, flow]
        copy_ch<<<eg(npix * 32), 256, 0, stream>>>(catD, 96, 64, ctx_in, 36, 0, B, 32, H, W);
        copy_ch<<<eg(HW2), 256, 0, stream>>>(flow, 2, 0, ctx_in, 36, 32, B, 2, H, W);
        conv(ctx_in, 36,  H, W, ctxW[0], ctxP0, 128, 0, H, W, 1, CTX_D[0], 1);
        conv(ctxP0,  128, H, W, ctxW[1], ctxP1, 128, 0, H, W, 1, CTX_D[1], 1);
        conv(ctxP1,  128, H, W, ctxW[2], ctxP0, 128, 0, H, W, 1, CTX_D[2], 1);
        conv(ctxP0,  128, H, W, ctxW[3], ctxP1, 96,  0, H, W, 1, CTX_D[3], 1);
        conv(ctxP1,  96,  H, W, ctxW[4], ctxP0, 64,  0, H, W, 1, CTX_D[4], 1);
        conv(ctxP0,  64,  H, W, ctxW[5], ctxP1, 32,  0, H, W, 1, CTX_D[5], 1);
        conv(ctxP1,  32,  H, W, ctxW[6], ctx_out, 2, 0, H, W, 1, CTX_D[6], 0);
        add_inplace<<<eg(HW2), 256, 0, stream>>>(flow, ctx_out, HW2);
    }

    // ---- final flow (NHWC) -> d_out (NCHW [B,2,64,208]) ----
    int nout = B * 2 * 64 * 208;
    nhwc_to_nchw<<<eg(nout), 256, 0, stream>>>(flow, (float*)d_out, B, 2, 64, 208);
}
